// DDSPHead_30081950941722
// MI455X (gfx1250) — matrix operand order repaired and checked by reading, not compiled, not run on hardware
//
#include <hip/hip_runtime.h>
#include <hip/hip_bf16.h>
#include <math.h>

typedef __attribute__((ext_vector_type(16))) _Float16 v16h;
typedef __attribute__((ext_vector_type(8)))  float    v8f;
typedef __attribute__((ext_vector_type(2)))  float    v2f;

#define SR_F     24000.0f
#define T_STEPS  400
#define NH       64
#define NB       65
#define CH       256
#define BATCH    4
#define HOP      256
#define NFFT     1024
#define NUP      102400            // T*HOP
#define NFRAMES  1600              // B*T
#define MPAD     144               // 129 outputs padded to 9 tiles of 16
#define NCOLF    1056              // 2*513 = 1026 padded to 33 pairs of 32
#define TWO_PI   6.283185307179586f

// ---------------- prep kernels ----------------

__global__ void k_prep_w(const float* __restrict__ w_h, const float* __restrict__ b_h,
                         const float* __restrict__ w_n, const float* __restrict__ b_n,
                         _Float16* __restrict__ Wh, float* __restrict__ biasF) {
  int tid = blockIdx.x * blockDim.x + threadIdx.x;
  if (tid >= MPAD * CH) return;
  int o = tid >> 8, c = tid & 255;
  float v = 0.f;
  if (o < 64)       v = w_h[o * CH + c];
  else if (o < 129) v = w_n[(o - 64) * CH + c];
  Wh[tid] = (_Float16)v;
  if (c == 0) {
    float bv = 0.f;
    if (o < 64)       bv = b_h[o];
    else if (o < 129) bv = b_n[o - 64];
    biasF[o] = bv;
  }
}

// zh pair-interleaved over k: element (k, n) at ((k>>1)*NFRAMES + n)*2 + (k&1)
__global__ void k_prep_z(const float* __restrict__ z, _Float16* __restrict__ zh) {
  int tid = blockIdx.x * blockDim.x + threadIdx.x;
  if (tid >= CH * NFRAMES) return;
  int k = tid / NFRAMES, n = tid % NFRAMES;
  int b = n / T_STEPS, t = n % T_STEPS;
  zh[((size_t)(k >> 1) * NFRAMES + n) * 2 + (k & 1)] =
      (_Float16)z[(b * CH + k) * T_STEPS + t];
}

__global__ void k_prep_frames(const float* __restrict__ noise, float* __restrict__ frames) {
  int tid = blockIdx.x * blockDim.x + threadIdx.x;
  if (tid >= NFRAMES * NFFT) return;
  int m = tid >> 10, nn = tid & 1023;
  int b = m / T_STEPS, t = m % T_STEPS;
  int s = t * HOP + nn - (NFFT / 2);
  float x = (s >= 0 && s < NUP) ? noise[b * NUP + s] : 0.f;
  float w = 0.5f * (1.f - __cosf((float)nn * (TWO_PI / (float)NFFT)));
  frames[tid] = x * w;
}

// Forward DFT matrix, pair-interleaved over k (k = time sample 0..1023).
__global__ void k_dft_fwd(float* __restrict__ dftF) {
  int tid = blockIdx.x * blockDim.x + threadIdx.x;
  if (tid >= NFFT * NCOLF) return;
  int k = tid / NCOLF, col = tid % NCOLF;
  float v = 0.f;
  if (col < 1026) {
    int f = col >> 1;
    int p = (k * f) & 1023;                      // exact phase mod N
    float ang = (float)p * (TWO_PI / (float)NFFT);
    v = (col & 1) ? -__sinf(ang) : __cosf(ang);  // X = sum x (cos - i sin)
  }
  dftF[((size_t)(k >> 1) * NCOLF + col) * 2 + (k & 1)] = v;
}

// Inverse DFT matrix, pair-interleaved over k (k = packed freq col 0..1055).
__global__ void k_dft_inv(float* __restrict__ dftI) {
  int tid = blockIdx.x * blockDim.x + threadIdx.x;
  if (tid >= NCOLF * NFFT) return;
  int k = tid / NFFT, n = tid % NFFT;
  float v = 0.f;
  if (k < 1026) {
    int f = k >> 1;
    int p = (f * n) & 1023;
    float ang = (float)p * (TWO_PI / (float)NFFT);
    float cf = (f == 0 || f == 512) ? 1.f : 2.f; // hermitian fold
    float scale = cf * (1.f / (float)NFFT);
    v = (k & 1) ? -scale * __sinf(ang) : scale * __cosf(ang);
  }
  dftI[((size_t)(k >> 1) * NFFT + n) * 2 + (k & 1)] = v;
}

// ---------------- feature GEMM (f16 WMMA 16x16x32) ----------------
// D(144 x 1600) = W(144 x 256) * Z(256 x 1600), postprocess to amp / Hmag.

__global__ void __launch_bounds__(256) k_gemm_feat(
    const _Float16* __restrict__ Wh, const float* __restrict__ biasF,
    const _Float16* __restrict__ zh, const float* __restrict__ f0,
    float* __restrict__ ampBuf, float* __restrict__ HmBuf) {
  int wave = (blockIdx.x * blockDim.x + threadIdx.x) >> 5;
  int lane = threadIdx.x & 31;
  const int NT = NFRAMES / 16;     // 100
  const int MT = MPAD / 16;        // 9
  if (wave >= MT * NT) return;
  int mt = wave / NT, nt = wave % NT;
  int n    = nt * 16 + (lane & 15);
  int mrow = mt * 16 + (lane & 15);
  int hi   = lane >> 4;            // 0 or 1
  v8f acc = {};
#pragma unroll
  for (int kb = 0; kb < CH; kb += 32) {
    union { v16h v; unsigned u[8]; } af, bf;
#pragma unroll
    for (int j = 0; j < 8; ++j) {
      // A halves h=2j,2j+1 -> k = 2*(j&3) + 8*hi + 16*(j>>2), k+1 (contiguous in W row)
      int ka = kb + 2 * (j & 3) + (hi << 3) + ((j >> 2) << 4);
      af.u[j] = *(const unsigned*)(Wh + mrow * CH + ka);
      // B halves h=2j,2j+1 -> k = 2*(j&3) + 8*hi + 16*(j>>2), k+1 (contiguous in paired zh)
      int kbn = kb + 2 * (j & 3) + (hi << 3) + ((j >> 2) << 4);
      bf.u[j] = *(const unsigned*)(zh + ((size_t)(kbn >> 1) * NFRAMES + n) * 2);
    }
    acc = __builtin_amdgcn_wmma_f32_16x16x32_f16(false, af.v, false, bf.v, (short)0, acc, false, false);
  }
  int bb = n / T_STEPS, t = n % T_STEPS;
  float f0v = f0[bb * T_STEPS + t];
#pragma unroll
  for (int r = 0; r < 8; ++r) {
    int o = mt * 16 + r + (hi << 3);
    float x = acc[r];
    if (o < 64) {
      x += biasF[o];
      float sp = (x > 20.f) ? x : log1pf(expf(x));     // softplus
      if (f0v * (float)(o + 1) >= (SR_F * 0.5f)) sp = 0.f;
      ampBuf[(bb * T_STEPS + t) * NH + o] = sp;
    } else if (o < 129) {
      x += biasF[o];
      float e = expf(x);
      if (e > 100.f) e = 100.f;
      HmBuf[(bb * T_STEPS + t) * NB + (o - 64)] = e;
    }
  }
}

// ---------------- phase scan (hierarchical, double) ----------------

__global__ void k_scanA(const float* __restrict__ f0, double* __restrict__ csbuf,
                        double* __restrict__ chunkSum) {
  int blk = blockIdx.x;
  int b = blk / 400, chunk = blk % 400;
  int tid = threadIdx.x;
  int s = chunk * 256 + tid;
  float pos = (float)s * (399.0f / 102399.0f);
  int i0 = (int)pos; if (i0 > 398) i0 = 398;
  float frac = pos - (float)i0;
  float fv = f0[b * T_STEPS + i0] * (1.f - frac) + f0[b * T_STEPS + i0 + 1] * frac;
  double v = (double)(fv / SR_F);
  __shared__ double sm[256];
  sm[tid] = v;
  __syncthreads();
  for (int off = 1; off < 256; off <<= 1) {
    double t = (tid >= off) ? sm[tid - off] : 0.0;
    __syncthreads();
    sm[tid] += t;
    __syncthreads();
  }
  csbuf[b * NUP + s] = sm[tid];
  if (tid == 255) chunkSum[blk] = sm[255];
}

__global__ void k_scanB(const double* __restrict__ chunkSum, double* __restrict__ chunkOff) {
  int b = threadIdx.x;
  if (b >= BATCH) return;
  double off = 0.0;
  for (int c = 0; c < 400; ++c) {
    chunkOff[b * 400 + c] = off;
    off += chunkSum[b * 400 + c];
  }
}

// ---------------- harmonic synthesis (Chebyshev recurrence) ----------------

__global__ void __launch_bounds__(256) k_periodic(
    const double* __restrict__ csbuf, const double* __restrict__ chunkOff,
    const float* __restrict__ ampBuf, float* __restrict__ periodicBuf) {
  int tid = blockIdx.x * blockDim.x + threadIdx.x;
  if (tid >= BATCH * NUP) return;
  int b = tid / NUP, s = tid % NUP;
  double cs = csbuf[tid] + chunkOff[b * 400 + (s >> 8)];
  double rev = cs - floor(cs);                 // exact: sin(2pi*h*cs) == sin(2pi*h*frac(cs))
  float theta = (float)(rev * (double)TWO_PI);
  float st, ct;
  __sincosf(theta, &st, &ct);
  float c2 = 2.f * ct;
  float pos = (float)s * (399.0f / 102399.0f);
  int i0 = (int)pos; if (i0 > 398) i0 = 398;
  float frac = pos - (float)i0;
  const float4* a0 = (const float4*)(ampBuf + (size_t)(b * T_STEPS + i0) * NH);
  const float4* a1 = a0 + (NH / 4);
  float skm1 = 0.f, sk = st, acc = 0.f;
#pragma unroll
  for (int q = 0; q < NH / 4; ++q) {
    float4 u = a0[q], w = a1[q];
    float av0 = u.x + (w.x - u.x) * frac;
    float av1 = u.y + (w.y - u.y) * frac;
    float av2 = u.z + (w.z - u.z) * frac;
    float av3 = u.w + (w.w - u.w) * frac;
    acc = fmaf(av0, sk, acc);
    float n1 = fmaf(c2, sk, -skm1); skm1 = sk; sk = n1;
    acc = fmaf(av1, sk, acc);
    float n2 = fmaf(c2, sk, -skm1); skm1 = sk; sk = n2;
    acc = fmaf(av2, sk, acc);
    float n3 = fmaf(c2, sk, -skm1); skm1 = sk; sk = n3;
    acc = fmaf(av3, sk, acc);
    float n4 = fmaf(c2, sk, -skm1); skm1 = sk; sk = n4;
  }
  periodicBuf[tid] = acc;
}

// ---------------- big f32 GEMM (WMMA 16x16x4 f32), 2x2 tiles per wave ----------------
// D(MxN) = A(MxK, row-major) * B(KxN, pair-interleaved over k).
// Software-pipelined: next K-step's fragments are fetched before the current WMMAs.

__global__ void __launch_bounds__(256) k_gemm_f32(
    const float* __restrict__ A, const float* __restrict__ Bp,
    float* __restrict__ D, int M, int N, int K) {
  int wave = (blockIdx.x * blockDim.x + threadIdx.x) >> 5;
  int lane = threadIdx.x & 31;
  int npair = N >> 5, mpair = M >> 5;
  if (wave >= mpair * npair) return;
  int mt = wave / npair, nt = wave % npair;
  int l15 = lane & 15;
  int m0 = mt * 32 + l15;
  int n0 = nt * 32 + l15;
  int hi = lane >> 4;
  int koff = hi << 1;                          // f32 A/B: k = v + 2*(lane>=16)
  v8f acc00 = {}, acc01 = {}, acc10 = {}, acc11 = {};

  // per-lane streaming pointers
  const float* aptr0 = A + (size_t)m0 * K + koff;
  const float* aptr1 = aptr0 + (size_t)16 * K;
  const float* bptr0 = Bp + ((size_t)hi * N + n0) * 2;   // (k=koff, n0) pair slot
  const float* bptr1 = bptr0 + 32;                        // n0+16
  const size_t bstep = (size_t)4 * N;                     // advance 4 k-rows (2 pairs)

  // prologue: load fragments for kb = 0
  v2f a0c = *(const v2f*)aptr0;
  v2f a1c = *(const v2f*)aptr1;
  v2f b0c = *(const v2f*)bptr0;
  v2f b1c = *(const v2f*)bptr1;

#pragma unroll 2
  for (int kb = 0; kb < K - 4; kb += 4) {
    aptr0 += 4; aptr1 += 4; bptr0 += bstep; bptr1 += bstep;
    v2f a0n = *(const v2f*)aptr0;              // prefetch next K-step
    v2f a1n = *(const v2f*)aptr1;
    v2f b0n = *(const v2f*)bptr0;
    v2f b1n = *(const v2f*)bptr1;
    acc00 = __builtin_amdgcn_wmma_f32_16x16x4_f32(false, a0c, false, b0c, (short)0, acc00, false, false);
    acc01 = __builtin_amdgcn_wmma_f32_16x16x4_f32(false, a0c, false, b1c, (short)0, acc01, false, false);
    acc10 = __builtin_amdgcn_wmma_f32_16x16x4_f32(false, a1c, false, b0c, (short)0, acc10, false, false);
    acc11 = __builtin_amdgcn_wmma_f32_16x16x4_f32(false, a1c, false, b1c, (short)0, acc11, false, false);
    a0c = a0n; a1c = a1n; b0c = b0n; b1c = b1n;
  }
  // epilogue: last K-step
  acc00 = __builtin_amdgcn_wmma_f32_16x16x4_f32(false, a0c, false, b0c, (short)0, acc00, false, false);
  acc01 = __builtin_amdgcn_wmma_f32_16x16x4_f32(false, a0c, false, b1c, (short)0, acc01, false, false);
  acc10 = __builtin_amdgcn_wmma_f32_16x16x4_f32(false, a1c, false, b0c, (short)0, acc10, false, false);
  acc11 = __builtin_amdgcn_wmma_f32_16x16x4_f32(false, a1c, false, b1c, (short)0, acc11, false, false);

  int hi8 = hi << 3;
#pragma unroll
  for (int r = 0; r < 8; ++r) {
    int mra = mt * 32 + r + hi8;
    int mrb = mra + 16;
    D[(size_t)mra * N + n0]      = acc00[r];
    D[(size_t)mra * N + n0 + 16] = acc01[r];
    D[(size_t)mrb * N + n0]      = acc10[r];
    D[(size_t)mrb * N + n0 + 16] = acc11[r];
  }
}

// ---------------- spectral multiply with band-upsampled H ----------------

__global__ void k_modspec(const float* __restrict__ Fbuf, const float* __restrict__ HmBuf,
                          float* __restrict__ Ymod) {
  int tid = blockIdx.x * blockDim.x + threadIdx.x;
  if (tid >= NFRAMES * 528) return;
  int m = tid / 528, f = tid % 528;
  if (f >= 513) {                              // zero pad cols 1026..1055
    Ymod[(size_t)m * NCOLF + 2 * f]     = 0.f;
    Ymod[(size_t)m * NCOLF + 2 * f + 1] = 0.f;
    return;
  }
  int i0 = f >> 3; if (i0 > 63) i0 = 63;       // pos = f*64/512 = f/8, clip like reference
  float frac = (float)f * 0.125f - (float)i0;
  float h0 = HmBuf[m * NB + i0];
  float h1 = HmBuf[m * NB + i0 + 1];
  float H = h0 * (1.f - frac) + h1 * frac;
  Ymod[(size_t)m * NCOLF + 2 * f]     = Fbuf[(size_t)m * NCOLF + 2 * f]     * H;
  Ymod[(size_t)m * NCOLF + 2 * f + 1] = Fbuf[(size_t)m * NCOLF + 2 * f + 1] * H;
}

// ---------------- overlap-add + normalize + final sum ----------------

__global__ void k_finalize(const float* __restrict__ yframes, const float* __restrict__ periodicBuf,
                           float* __restrict__ out) {
  int tid = blockIdx.x * blockDim.x + threadIdx.x;
  if (tid >= BATCH * NUP) return;
  int b = tid / NUP, s = tid % NUP;
  int idx = s + (NFFT / 2);
  int tbase = idx >> 8;
  float sum = 0.f, wsum = 0.f;
#pragma unroll
  for (int dt = 0; dt < 4; ++dt) {
    int t = tbase - dt;
    if (t < 0 || t > 399) continue;
    int nn = idx - t * HOP;                    // guaranteed 0..1023
    float w = 0.5f * (1.f - __cosf((float)nn * (TWO_PI / (float)NFFT)));
    sum  += yframes[(size_t)(b * T_STEPS + t) * NFFT + nn] * w;
    wsum += w * w;
  }
  float ap  = sum / (wsum + 1e-8f);
  float per = periodicBuf[tid];
  out[tid]                    = per + ap;      // y
  out[BATCH * NUP + tid]      = per;           // periodic
  out[2 * BATCH * NUP + tid]  = ap;            // aperiodic
}

// ---------------- launch ----------------

extern "C" void kernel_launch(void* const* d_in, const int* in_sizes, int n_in,
                              void* d_out, int out_size, void* d_ws, size_t ws_size,
                              hipStream_t stream) {
  (void)in_sizes; (void)n_in; (void)out_size; (void)ws_size;
  const float* z     = (const float*)d_in[0];
  const float* f0    = (const float*)d_in[1];
  const float* w_h   = (const float*)d_in[2];
  const float* b_h   = (const float*)d_in[3];
  const float* w_n   = (const float*)d_in[4];
  const float* b_n   = (const float*)d_in[5];
  const float* noise = (const float*)d_in[6];
  float* out = (float*)d_out;

  char* base = (char*)d_ws;
  size_t off = 0;
  auto take = [&](size_t bytes) -> char* {
    char* p = base + off;
    off = (off + bytes + 255) & ~(size_t)255;
    return p;
  };
  _Float16* Wh        = (_Float16*)take((size_t)MPAD * CH * 2);
  float*    biasF     = (float*)   take((size_t)MPAD * 4);
  _Float16* zh        = (_Float16*)take((size_t)CH * NFRAMES * 2);
  float*    ampBuf    = (float*)   take((size_t)NFRAMES * NH * 4);
  float*    HmBuf     = (float*)   take((size_t)NFRAMES * NB * 4);
  double*   csbuf     = (double*)  take((size_t)BATCH * NUP * 8);
  double*   chunkSum  = (double*)  take((size_t)BATCH * 400 * 8);
  double*   chunkOff  = (double*)  take((size_t)BATCH * 400 * 8);
  float*    periodicB = (float*)   take((size_t)BATCH * NUP * 4);
  float*    frames    = (float*)   take((size_t)NFRAMES * NFFT * 4);
  float*    dftF      = (float*)   take((size_t)NFFT * NCOLF * 4);
  float*    dftI      = (float*)   take((size_t)NCOLF * NFFT * 4);
  float*    Fbuf      = (float*)   take((size_t)NFRAMES * NCOLF * 4);
  float*    Ymod      = (float*)   take((size_t)NFRAMES * NCOLF * 4);
  float*    yframes   = (float*)   take((size_t)NFRAMES * NFFT * 4);

  // prep
  k_prep_w     <<<(MPAD * CH + 255) / 256, 256, 0, stream>>>(w_h, b_h, w_n, b_n, Wh, biasF);
  k_prep_z     <<<(CH * NFRAMES + 255) / 256, 256, 0, stream>>>(z, zh);
  k_prep_frames<<<(NFRAMES * NFFT + 255) / 256, 256, 0, stream>>>(noise, frames);
  k_dft_fwd    <<<(NFFT * NCOLF + 255) / 256, 256, 0, stream>>>(dftF);
  k_dft_inv    <<<(NCOLF * NFFT + 255) / 256, 256, 0, stream>>>(dftI);

  // feature GEMM (WMMA f16): 9*100 tiles, 1 wave each
  {
    int waves = (MPAD / 16) * (NFRAMES / 16);
    k_gemm_feat<<<(waves * 32 + 255) / 256, 256, 0, stream>>>(Wh, biasF, zh, f0, ampBuf, HmBuf);
  }

  // phase scan + harmonic synthesis
  k_scanA   <<<BATCH * 400, 256, 0, stream>>>(f0, csbuf, chunkSum);
  k_scanB   <<<1, BATCH, 0, stream>>>(chunkSum, chunkOff);
  k_periodic<<<(BATCH * NUP + 255) / 256, 256, 0, stream>>>(csbuf, chunkOff, ampBuf, periodicB);

  // forward DFT GEMM (WMMA f32, 2x2 tiles): (1600x1024)*(1024x1056)
  {
    int waves = (NFRAMES / 32) * (NCOLF / 32);   // 50*33
    k_gemm_f32<<<(waves * 32 + 255) / 256, 256, 0, stream>>>(frames, dftF, Fbuf, NFRAMES, NCOLF, NFFT);
  }

  k_modspec<<<(NFRAMES * 528 + 255) / 256, 256, 0, stream>>>(Fbuf, HmBuf, Ymod);

  // inverse DFT GEMM (WMMA f32, 2x2 tiles): (1600x1056)*(1056x1024)
  {
    int waves = (NFRAMES / 32) * (NFFT / 32);    // 50*32
    k_gemm_f32<<<(waves * 32 + 255) / 256, 256, 0, stream>>>(Ymod, dftI, yframes, NFRAMES, NFFT, NCOLF);
  }

  k_finalize<<<(BATCH * NUP + 255) / 256, 256, 0, stream>>>(yframes, periodicB, out);
}
